// CustomLSTM_71021579206862
// MI455X (gfx1250) — matrix operand order repaired and checked by reading, not compiled, not run on hardware
//
#include <hip/hip_runtime.h>
#include <hip/hip_bf16.h>

#define N_CLASSES  4
#define N_FEATURES 13
#define N_HIDDEN   13
#define N_STEPS    100
#define FORGET_BIAS 1.0f

typedef __attribute__((ext_vector_type(16))) _Float16 v16h;
typedef __attribute__((ext_vector_type(8)))  _Float16 v8h;
typedef __attribute__((ext_vector_type(8)))  float    v8f;

#define LDS_FENCE() asm volatile("s_wait_dscnt 0" ::: "memory")

__device__ __forceinline__ v8f wmma16(v16h a, v16h b, v8f c) {
  // D = A(16x32 f16) * B(32x16 f16) + C(16x16 f32)
  return __builtin_amdgcn_wmma_f32_16x16x32_f16(
      /*neg_a=*/false, a, /*neg_b=*/false, b,
      /*c_mod=*/(short)0, c, /*reuse_a=*/false, /*reuse_b=*/false);
}

__device__ __forceinline__ v8f splat8(float v) {
  return (v8f){v, v, v, v, v, v, v, v};
}

__device__ __forceinline__ float sigf(float x) {
  // sigmoid via single TANH trans op (CDNA5 v_tanh_f32 pipe)
  return __builtin_fmaf(0.5f, tanhf(0.5f * x), 0.5f);
}

__global__ __launch_bounds__(256)
void lstm_fused_kernel(const float* __restrict__ xin,
                       const float* __restrict__ w_hidden,
                       const float* __restrict__ b_hidden,
                       const float* __restrict__ k0,
                       const float* __restrict__ bias0,
                       const float* __restrict__ k1,
                       const float* __restrict__ bias1,
                       const float* __restrict__ w_output,
                       const float* __restrict__ b_output,
                       float* __restrict__ out,
                       int Btotal) {
  // One wave = one 16-row batch tile. 8 waves / block, fully independent.
  const int lane = threadIdx.x & 31;
  const int w    = threadIdx.x >> 5;
  const int r0   = (blockIdx.x * 8 + w) * 16;   // batch row base of this tile
  if (r0 >= Btotal) return;                      // wave-uniform: EXEC stays all-1s

  const int col   = lane & 15;                   // C/D + B-operand column
  const int half  = (lane < 16) ? 0 : 1;
  const int f0    = half * 8;                    // A-layout contiguous half offset
  const int rowM  = lane & 15;                   // A-layout row owned by this lane
  const int cdoff = half * 8;                    // C/D: M = r + cdoff

  // Per-wave LDS tiles: 16x16 f16, row-major (stride 16 halves = 32B)
  __shared__ __align__(16) _Float16 s_x [8][256];
  __shared__ __align__(16) _Float16 s_h0[8][256];
  __shared__ __align__(16) _Float16 s_h1[8][256];
  __shared__ __align__(16) float    s_fin[8][256];

  // ---- Build loop-invariant B fragments in registers (one-time) ----
  // Lane layout (dense 32x16 f16 B): element e<8 holds K=8*half+e,
  // element e>=8 holds K=16+8*half+(e-8).
  const int baseK = half * 16;

  v16h bp;                                       // w_hidden padded 13x13 -> 32x16
  #pragma unroll
  for (int e = 0; e < 16; ++e) {
    const int k = (e < 8) ? (f0 + e) : (16 + f0 + e - 8);
    float v = 0.f;
    if (col < N_HIDDEN && k < N_FEATURES) v = w_hidden[k * N_HIDDEN + col];
    bp[e] = (_Float16)v;
  }

  v16h b0[4], b1[4];                             // gate-padded LSTM kernels 26x52 -> 32x64
  #pragma unroll
  for (int g = 0; g < 4; ++g) {
    #pragma unroll
    for (int e = 0; e < 16; ++e) {
      const int k = (e < 8) ? (f0 + e) : (16 + f0 + e - 8);
      // K 0..12 -> x rows 0..12 ; K 16..28 -> h rows 13..25 ; else zero pad
      int row = -1;
      if (k < 13) row = k;
      else if (k >= 16 && k < 29) row = 13 + (k - 16);
      float v0 = 0.f, v1 = 0.f;
      if (col < N_HIDDEN && row >= 0) {
        v0 = k0[row * (4 * N_HIDDEN) + g * N_HIDDEN + col];
        v1 = k1[row * (4 * N_HIDDEN) + g * N_HIDDEN + col];
      }
      b0[g][e] = (_Float16)v0;
      b1[g][e] = (_Float16)v1;
    }
  }

  // Per-lane bias scalars (broadcast over M inside a fragment column)
  const bool cok = (col < N_HIDDEN);
  const float bh  = cok ? b_hidden[col] : 0.f;
  const float bi0 = cok ? bias0[0 * N_HIDDEN + col] : 0.f;
  const float bj0 = cok ? bias0[1 * N_HIDDEN + col] : 0.f;
  const float bf0 = cok ? bias0[2 * N_HIDDEN + col] : 0.f;
  const float bo0 = cok ? bias0[3 * N_HIDDEN + col] : 0.f;
  const float bi1 = cok ? bias1[0 * N_HIDDEN + col] : 0.f;
  const float bj1 = cok ? bias1[1 * N_HIDDEN + col] : 0.f;
  const float bf1 = cok ? bias1[2 * N_HIDDEN + col] : 0.f;
  const float bo1 = cok ? bias1[3 * N_HIDDEN + col] : 0.f;

  // Zero h0/h1 tiles (initial hidden state)
  #pragma unroll
  for (int i = 0; i < 8; ++i) {
    s_h0[w][lane * 8 + i] = (_Float16)0.f;
    s_h1[w][lane * 8 + i] = (_Float16)0.f;
  }
  LDS_FENCE();

  // Cell states live in f32 C/D fragments for all 100 steps
  v8f c0 = splat8(0.f), c1 = splat8(0.f), hl = splat8(0.f);

  const int    brow  = min(r0 + rowM, Btotal - 1);
  const size_t xbase = (size_t)brow * (N_STEPS * N_FEATURES);

  // Per-lane constant: which of the 8 feature slots are real (f0+e < 13),
  // and the clamped (always in-bounds) load offsets. Branch-free loads.
  bool  fvalid[8];
  int   foff[8];
  #pragma unroll
  for (int e = 0; e < 8; ++e) {
    const int f  = f0 + e;
    fvalid[e] = (f < N_FEATURES);
    foff[e]   = (f < N_FEATURES) ? f : (N_FEATURES - 1);  // clamp: stays in-bounds
  }

  // A-fragment of raw features for step t (x in K 0..12, rest zero).
  // Streamed non-temporally: 170MB read-once, keep the 192MB L2 clean.
  // Loads are unconditional (clamped address) + cndmask zeroing: no exec churn.
  auto loadAx = [&](int t) -> v16h {
    v16h ax;
    const float* p = xin + xbase + (size_t)t * N_FEATURES;
    #pragma unroll
    for (int e = 0; e < 8; ++e) {
      const float xv = __builtin_nontemporal_load(p + foff[e]);
      ax[e] = (_Float16)(fvalid[e] ? xv : 0.f);
    }
    #pragma unroll
    for (int e = 8; e < 16; ++e) ax[e] = (_Float16)0.f;
    return ax;
  };

  v16h axc = loadAx(0);

  for (int t = 0; t < N_STEPS; ++t) {
    // Prefetch next timestep's features (clamped index: straight-line code)
    v16h axn = loadAx(min(t + 1, N_STEPS - 1));

    // ---- input projection: relu(x @ w_hidden + b_hidden) ----
    v8f zp = wmma16(axc, bp, splat8(0.f));
    #pragma unroll
    for (int r = 0; r < 8; ++r) {
      float v = fmaxf(zp[r] + bh, 0.f);
      s_x[w][(r + cdoff) * 16 + col] = (_Float16)(cok ? v : 0.f);
    }
    LDS_FENCE();

    // ---- LSTM cell 0: A = [xproj | h0] ----
    {
      v8h lo = *(const v8h*)&s_x [w][rowM * 16 + f0];
      v8h hi = *(const v8h*)&s_h0[w][rowM * 16 + f0];
      v16h a0 = __builtin_shufflevector(lo, hi, 0,1,2,3,4,5,6,7,8,9,10,11,12,13,14,15);
      v8f zi = wmma16(a0, b0[0], splat8(bi0));
      v8f zj = wmma16(a0, b0[1], splat8(bj0));
      v8f zf = wmma16(a0, b0[2], splat8(bf0));
      v8f zo = wmma16(a0, b0[3], splat8(bo0));
      #pragma unroll
      for (int r = 0; r < 8; ++r) {
        float ig = sigf(zi[r]);
        float jg = tanhf(zj[r]);
        float fg = sigf(zf[r] + FORGET_BIAS);
        float og = sigf(zo[r]);
        float cn = c0[r] * fg + ig * jg;
        c0[r] = cn;
        float hh = tanhf(cn) * og;
        s_h0[w][(r + cdoff) * 16 + col] = (_Float16)(cok ? hh : 0.f);
      }
    }
    LDS_FENCE();

    // ---- LSTM cell 1: A = [h0 | h1] ----
    {
      v8h lo = *(const v8h*)&s_h0[w][rowM * 16 + f0];
      v8h hi = *(const v8h*)&s_h1[w][rowM * 16 + f0];
      v16h a1 = __builtin_shufflevector(lo, hi, 0,1,2,3,4,5,6,7,8,9,10,11,12,13,14,15);
      v8f zi = wmma16(a1, b1[0], splat8(bi1));
      v8f zj = wmma16(a1, b1[1], splat8(bj1));
      v8f zf = wmma16(a1, b1[2], splat8(bf1));
      v8f zo = wmma16(a1, b1[3], splat8(bo1));
      #pragma unroll
      for (int r = 0; r < 8; ++r) {
        float ig = sigf(zi[r]);
        float jg = tanhf(zj[r]);
        float fg = sigf(zf[r] + FORGET_BIAS);
        float og = sigf(zo[r]);
        float cn = c1[r] * fg + ig * jg;
        c1[r] = cn;
        float hh = tanhf(cn) * og;
        hl[r] = hh;                                    // keep final h1 in f32
        s_h1[w][(r + cdoff) * 16 + col] = (_Float16)(cok ? hh : 0.f);
      }
    }
    LDS_FENCE();

    axc = axn;
  }

  // ---- output projection: h1_final @ w_output + b_output ----
  #pragma unroll
  for (int r = 0; r < 8; ++r)
    s_fin[w][(r + cdoff) * 16 + col] = cok ? hl[r] : 0.f;
  LDS_FENCE();

  const int orow  = lane & 15;           // tile row
  const int opair = lane >> 4;           // class pair {0,1} or {2,3}
  float acc0 = b_output[opair * 2 + 0];
  float acc1 = b_output[opair * 2 + 1];
  #pragma unroll
  for (int n = 0; n < N_HIDDEN; ++n) {
    const float hv = s_fin[w][orow * 16 + n];
    acc0 += hv * w_output[n * N_CLASSES + opair * 2 + 0];
    acc1 += hv * w_output[n * N_CLASSES + opair * 2 + 1];
  }
  const int ob = r0 + orow;
  if (ob < Btotal) {
    out[(size_t)ob * N_CLASSES + opair * 2 + 0] = acc0;
    out[(size_t)ob * N_CLASSES + opair * 2 + 1] = acc1;
  }
}

extern "C" void kernel_launch(void* const* d_in, const int* in_sizes, int n_in,
                              void* d_out, int out_size, void* d_ws, size_t ws_size,
                              hipStream_t stream) {
  const float* x        = (const float*)d_in[0];
  const float* w_hidden = (const float*)d_in[1];
  const float* b_hidden = (const float*)d_in[2];
  const float* k0       = (const float*)d_in[3];
  const float* bias0    = (const float*)d_in[4];
  const float* k1       = (const float*)d_in[5];
  const float* bias1    = (const float*)d_in[6];
  const float* w_out    = (const float*)d_in[7];
  const float* b_out    = (const float*)d_in[8];
  float* out            = (float*)d_out;

  const int B = in_sizes[0] / (N_STEPS * N_FEATURES);
  const int rowsPerBlock = 8 * 16;               // 8 waves x 16-row tiles
  const int grid = (B + rowsPerBlock - 1) / rowsPerBlock;

  lstm_fused_kernel<<<grid, 256, 0, stream>>>(
      x, w_hidden, b_hidden, k0, bias0, k1, bias1, w_out, b_out, out, B);
}
